// HybridGPT_16793322127765
// MI455X (gfx1250) — matrix operand order repaired and checked by reading, not compiled, not run on hardware
//
#include <hip/hip_runtime.h>
#include <hip/hip_bf16.h>
#include <stdint.h>
#include <math.h>

#define DIM   1024
#define STATE 1024
#define SELH  128
#define NE    4
#define NB    2
#define NS    2048
#define NTOK  (NB*NS)            // 4096 tokens
#define NGRP  (NB*NE)            // 8 (batch,expert) groups
#define CAP   (NTOK + NGRP*64)   // 4608 padded rows (each group padded to x64)
#define MT    (CAP/64)           // 72 row tiles

typedef unsigned short u16;
typedef unsigned int   u32;
typedef __attribute__((ext_vector_type(16))) __bf16 v16bf;
typedef __attribute__((ext_vector_type(8)))  float  v8f;
typedef unsigned int u32x4 __attribute__((ext_vector_type(4)));
typedef unsigned int v4u  __attribute__((ext_vector_type(4)));
typedef int          v8i  __attribute__((ext_vector_type(8)));
typedef int          v4i  __attribute__((ext_vector_type(4)));

__device__ __forceinline__ u16 f2bf(float f) {
  union { float f; u32 u; } v; v.f = f;
  u32 r = v.u + 0x7fffu + ((v.u >> 16) & 1u);   // round-to-nearest-even
  return (u16)(r >> 16);
}
__device__ __forceinline__ float bf2f(u16 h) {
  union { u32 u; float f; } v; v.u = ((u32)h) << 16;
  return v.f;
}
__device__ __forceinline__ float sigmoidf_(float x) { return 1.0f / (1.0f + expf(-x)); }

// --------------------------------------------------------------- TDM 2D tile
// D# per CDNA5 ISA ch.8: group0 = {count|flags, lds_addr, global_addr, type},
// group1 = {data_size, tensor_dim0/1, tile_dim0/1, dim0_stride}. 2-byte elems.
// This toolchain exposes the 6-arg builtin: (g0, g1, g2, g3, g4, cpol).
__device__ __forceinline__ void tdm_load_2d(u32 lds_off, const void* gptr,
                                            u32 dim0, u32 dim1, u32 stride,
                                            u32 tile0, u32 tile1) {
  unsigned long long ga = (unsigned long long)gptr;
  v4u g0;
  g0[0] = 1u;                                            // count=1 (valid user D#)
  g0[1] = lds_off;                                       // lds_addr (bytes)
  g0[2] = (u32)(ga & 0xffffffffu);                       // global_addr[31:0]
  g0[3] = (u32)((ga >> 32) & 0x1ffffffu) | (2u << 30);   // addr[56:32] | type=2
  v8i g1;
  g1[0] = (int)(1u << 16);                               // data_size=2B; no mcast/pad/iter
  g1[1] = (int)((dim0 & 0xffffu) << 16);                 // tensor_dim0[15:0] -> bits 63:48
  g1[2] = (int)((dim0 >> 16) | ((dim1 & 0xffffu) << 16));// dim0 hi16 | dim1 lo16
  g1[3] = (int)((dim1 >> 16) | (tile0 << 16));           // dim1 hi16 | tile_dim0
  g1[4] = (int)(tile1 & 0xffffu);                        // tile_dim1 (tile_dim2=0)
  g1[5] = (int)stride;                                   // tensor_dim0_stride[31:0]
  g1[6] = 0;
  g1[7] = 0;
  v4i z4 = {0, 0, 0, 0};                                 // groups 2/3 unused (2-D)
  v8i z8 = {0, 0, 0, 0, 0, 0, 0, 0};
  __builtin_amdgcn_tensor_load_to_lds(g0, g1, z4, z4, z8, 0);
}

// ---------------------------------------------------------------- f32 -> bf16
__global__ void cvt_bf16(const float* __restrict__ src, u16* __restrict__ dst, int n) {
  for (int i = blockIdx.x * blockDim.x + threadIdx.x; i < n; i += gridDim.x * blockDim.x)
    dst[i] = f2bf(src[i]);
}

// ------------------------------------------- routes + stable group sort (1 block)
__global__ void route_build(const int* __restrict__ tok, int* __restrict__ perm,
                            int* __restrict__ meta /* cnt[8], off[9], tileExp[MT] */) {
  __shared__ int sroute[NTOK];
  const int tid = threadIdx.x;
  for (int t = tid; t < NTOK; t += blockDim.x) {
    u32 h = (u32)tok[t];
    h ^= h >> 16; h *= 2246822507u; h ^= h >> 13; h *= 3266489909u; h ^= h >> 16;
    sroute[t] = (int)(h & (u32)(NE - 1));   // % 4
  }
  __syncthreads();
  if (tid == 0) {
    int cnt[NGRP]; int off[NGRP + 1]; int pos[NGRP];
    for (int g = 0; g < NGRP; ++g) cnt[g] = 0;
    for (int t = 0; t < NTOK; ++t) cnt[(t >> 11) * NE + sroute[t]]++;
    off[0] = 0;
    for (int g = 0; g < NGRP; ++g) off[g + 1] = off[g] + ((cnt[g] + 63) & ~63);
    for (int r = 0; r < CAP; ++r) perm[r] = -1;
    for (int g = 0; g < NGRP; ++g) pos[g] = off[g];
    for (int t = 0; t < NTOK; ++t) {            // time-ordered within each group
      int g = (t >> 11) * NE + sroute[t];
      perm[pos[g]++] = t;
    }
    for (int g = 0; g < NGRP; ++g) meta[g] = cnt[g];
    for (int g = 0; g <= NGRP; ++g) meta[8 + g] = off[g];
    for (int tile = 0; tile < MT; ++tile) {
      int e = 0, rb = tile * 64;
      for (int g = 0; g < NGRP; ++g)
        if (rb >= off[g] && rb < off[g + 1]) e = g & (NE - 1);
      meta[17 + tile] = e;
    }
  }
}

// ------------------------------------------------- resid mix + rmsnorm -> bf16
__global__ void mix_rmsnorm(const float* __restrict__ x, const float* __restrict__ x0,
                            const float* __restrict__ mix, float* __restrict__ xm,
                            u16* __restrict__ xs) {
  const int t = blockIdx.x, tid = threadIdx.x;
  float4 a  = ((const float4*)(x  + (size_t)t * DIM))[tid];
  float4 b  = ((const float4*)(x0 + (size_t)t * DIM))[tid];
  float4 m0 = ((const float4*)mix)[tid];
  float4 m1 = ((const float4*)(mix + DIM))[tid];
  float4 v;
  v.x = m0.x * a.x + m1.x * b.x;
  v.y = m0.y * a.y + m1.y * b.y;
  v.z = m0.z * a.z + m1.z * b.z;
  v.w = m0.w * a.w + m1.w * b.w;
  ((float4*)(xm + (size_t)t * DIM))[tid] = v;
  float ss = v.x * v.x + v.y * v.y + v.z * v.z + v.w * v.w;
  __shared__ float red[256];
  red[tid] = ss; __syncthreads();
  for (int s = 128; s > 0; s >>= 1) { if (tid < s) red[tid] += red[tid + s]; __syncthreads(); }
  float r = rsqrtf(red[0] * (1.0f / DIM) + 1e-6f);
  uint2 pk;
  pk.x = (u32)f2bf(v.x * r) | ((u32)f2bf(v.y * r) << 16);
  pk.y = (u32)f2bf(v.z * r) | ((u32)f2bf(v.w * r) << 16);
  ((uint2*)(xs + (size_t)t * DIM))[tid] = pk;
}

__global__ void rmsnorm2(const float* __restrict__ x2, u16* __restrict__ xn) {
  const int t = blockIdx.x, tid = threadIdx.x;
  float4 v = ((const float4*)(x2 + (size_t)t * DIM))[tid];
  float ss = v.x * v.x + v.y * v.y + v.z * v.z + v.w * v.w;
  __shared__ float red[256];
  red[tid] = ss; __syncthreads();
  for (int s = 128; s > 0; s >>= 1) { if (tid < s) red[tid] += red[tid + s]; __syncthreads(); }
  float r = rsqrtf(red[0] * (1.0f / DIM) + 1e-6f);
  uint2 pk;
  pk.x = (u32)f2bf(v.x * r) | ((u32)f2bf(v.y * r) << 16);
  pk.y = (u32)f2bf(v.z * r) | ((u32)f2bf(v.w * r) << 16);
  ((uint2*)(xn + (size_t)t * DIM))[tid] = pk;
}

// ------------------- gather tokens into padded layout via async LDS DMA path
__global__ void gather_rows(const int* __restrict__ perm, const u16* __restrict__ xs,
                            u16* __restrict__ Xg) {
  __shared__ __align__(16) u16 buf[DIM];
  const int r = blockIdx.x, tid = threadIdx.x;
  const int t = perm[r];                               // uniform per block
  const u32 lds = ((u32)(uintptr_t)(&buf[0])) + (u32)(tid * 8);
  if (t >= 0) {
    const u16* src = xs + (size_t)t * DIM + tid * 4;
    asm volatile("global_load_async_to_lds_b64 %0, %1, off"
                 :: "v"(lds), "v"(src) : "memory");
  } else {
    uint2 z; z.x = 0u; z.y = 0u;
    ((uint2*)buf)[tid] = z;
  }
  asm volatile("s_wait_asynccnt 0x0" ::: "memory");
  asm volatile("s_wait_dscnt 0x0" ::: "memory");
  u16* dst = Xg + (size_t)r * DIM + tid * 4;
  asm volatile("global_store_async_from_lds_b64 %0, %1, off"
               :: "v"(dst), "v"(lds) : "memory");
  asm volatile("s_wait_asynccnt 0x0" ::: "memory");
}

// ---------------------------------- diagonal SSM scan: parallel over (group, k)
__global__ void ssm_scan(const u16* __restrict__ Ug, const u16* __restrict__ Sel,
                         const float* __restrict__ d_param, const int* __restrict__ meta,
                         u16* __restrict__ Yg) {
  const int g = blockIdx.x >> 2;
  const int k = ((blockIdx.x & 3) << 8) + threadIdx.x;
  const int e = g & (NE - 1);
  const int cnt  = meta[g];
  const int base = meta[8 + g];
  const float d = d_param[e * STATE + k];
  float h = 0.0f;
  for (int t = 0; t < cnt; ++t) {
    const size_t r  = (size_t)(base + t);
    const float u   = bf2f(Ug[r * STATE + k]);
    const size_t s4 = r * (size_t)(4 * STATE);
    const float a   = bf2f(Sel[s4 + k]);
    const float b   = bf2f(Sel[s4 + STATE + k]);
    const float c   = bf2f(Sel[s4 + 2 * STATE + k]);
    const float gt  = bf2f(Sel[s4 + 3 * STATE + k]);
    h = a * h + b * u;                       // routed step: mask == 1
    Yg[r * STATE + k] = f2bf(c * h + d * gt * u);
  }
}

// ------------------------------------------------------------- WMMA bf16 GEMM
// C(64x64 per block) = A[M,K](bf16,row-major) x B[K,N](bf16,row-major), f32 acc.
// A tile staged by the Tensor Data Mover (one TDM issue by wave 0, TENSORcnt);
// B tile staged transposed by all threads so K is LDS-contiguous per column.
// 8 waves: 4 row-bands x 2 col-bands, each wave = two 16x16x32 WMMA chains.
// EPI: 0=store bf16  1=silu bf16  2=sel activations bf16  3=scatter x2 += s*y
//      4=relu^2 bf16  5=out = resid + s*acc (f32)
template<int EPI>
__global__ __launch_bounds__(256)
void gemm_wmma(const u16* __restrict__ A, int lda, int Mrows,
               const u16* __restrict__ Bw, int N, int K, long long expStride,
               const int* __restrict__ tileExpert,
               const int* __restrict__ perm,
               u16* __restrict__ outBf, float* __restrict__ outF,
               const float* __restrict__ scale, float* __restrict__ resid) {
  __shared__ __align__(16) u16 As[64 * 32];    // [row][k]   (TDM destination)
  __shared__ __align__(16) u16 Bs[64 * 32];    // transposed: [n][k]
  const int tid  = threadIdx.x;
  const int lane = tid & 31;
  const int wave = tid >> 5;
  const int wm = (wave & 3) << 4;              // 16-row band
  const int wn = (wave >> 2) << 5;             // 32-col band
  const int row0 = blockIdx.y << 6;
  const int col0 = blockIdx.x << 6;

  const u16* Bbase = Bw;
  if (tileExpert) Bbase += (long long)tileExpert[blockIdx.y] * expStride;

  const int brow = tid >> 3, bseg = tid & 7;   // B stage: 32 k x 8 x (8 bf16)
  const u32 ldsA = (u32)(uintptr_t)(&As[0]);   // LDS byte address for TDM
  const u16* gaTile = A + (size_t)row0 * lda;  // A tile row 0 (k advances)
  const u16* gbBase = Bbase + (size_t)brow * N + col0 + bseg * 8;

  v8f acc0 = {};
  v8f acc1 = {};

  for (int k0 = 0; k0 < K; k0 += 32) {
    __syncthreads();                            // prev tile consumed
    if (wave == 0) {
      // TDM: 64x32 bf16 tile, rows strided by lda, packed into As (TENSORcnt)
      tdm_load_2d(ldsA, gaTile + k0, (u32)K, (u32)Mrows, (u32)lda, 32u, 64u);
    }
    // B tile: load row-major, store transposed so K is contiguous per column
    const u16* gb = gbBase + (size_t)k0 * N;
    union { u32x4 q; u16 h[8]; } bb;
    bb.q = *(const u32x4*)gb;
#pragma unroll
    for (int j = 0; j < 8; ++j)
      Bs[(bseg * 8 + j) * 32 + brow] = bb.h[j];
    if (k0 + 32 < K) {                          // prefetch next K tile
      __builtin_prefetch(gaTile + (size_t)(tid >> 2) * lda + k0 + 32, 0, 1);
      __builtin_prefetch(gb + (size_t)32 * N, 0, 1);
    }
    if (wave == 0) __builtin_amdgcn_s_wait_tensorcnt(0);
    __syncthreads();

    // A fragment (ISA 7.12.2): lanes 0-15 -> K 0..7 & 16..23, lanes 16-31 -> K 8..15 & 24..31
    const int am = wm + (lane & 15);
    const int kq = (lane >> 4) << 3;
    union { u32x4 q[2]; v16bf v; } af, bf0, bf1;
    af.q[0] = *(const u32x4*)&As[am * 32 + kq];
    af.q[1] = *(const u32x4*)&As[am * 32 + 16 + kq];
    // B fragment: same K order as A (lanes 0-15 -> K 0..7 & 16..23, lanes 16-31 -> K 8..15 & 24..31)
    const int kh = (lane >> 4) << 4;
    int n = wn + (lane & 15);
    bf0.q[0] = *(const u32x4*)&Bs[n * 32 + kq];
    bf0.q[1] = *(const u32x4*)&Bs[n * 32 + 16 + kq];
    n += 16;
    bf1.q[0] = *(const u32x4*)&Bs[n * 32 + kq];
    bf1.q[1] = *(const u32x4*)&Bs[n * 32 + 16 + kq];

    acc0 = __builtin_amdgcn_wmma_f32_16x16x32_bf16(false, af.v, false, bf0.v,
                                                   (short)0, acc0, false, false);
    acc1 = __builtin_amdgcn_wmma_f32_16x16x32_bf16(false, af.v, false, bf1.v,
                                                   (short)0, acc1, false, false);
  }

  // C layout: VGPR r -> M = r (+8 for lanes 16-31), N = lane&15
  const int rb = row0 + wm + ((lane >> 4) << 3);
  const int cb = col0 + wn + (lane & 15);
#pragma unroll
  for (int r = 0; r < 8; ++r) {
    const int row = rb + r;
    const float v0 = acc0[r], v1 = acc1[r];
    if (EPI == 3) {
      const int t = perm[row];
      if (t >= 0) {
        const size_t i0 = (size_t)t * N + cb;
        resid[i0]      += scale[cb]      * v0;
        resid[i0 + 16] += scale[cb + 16] * v1;
      }
    } else {
      const size_t i0 = (size_t)row * N + cb;
      if (EPI == 0) { outBf[i0] = f2bf(v0); outBf[i0 + 16] = f2bf(v1); }
      else if (EPI == 1) {
        outBf[i0]      = f2bf(v0 * sigmoidf_(v0));
        outBf[i0 + 16] = f2bf(v1 * sigmoidf_(v1));
      } else if (EPI == 2) {
        const int b0 = cb >> 10, b1 = (cb + 16) >> 10;
        outBf[i0]      = f2bf((b0 == 0 || b0 == 3) ? sigmoidf_(v0) : tanhf(v0));
        outBf[i0 + 16] = f2bf((b1 == 0 || b1 == 3) ? sigmoidf_(v1) : tanhf(v1));
      } else if (EPI == 4) {
        const float r0 = v0 > 0.0f ? v0 : 0.0f, r1 = v1 > 0.0f ? v1 : 0.0f;
        outBf[i0] = f2bf(r0 * r0); outBf[i0 + 16] = f2bf(r1 * r1);
      } else if (EPI == 5) {
        outF[i0]      = resid[i0]      + scale[cb]    * v0;
        outF[i0 + 16] = resid[i0 + 16] + scale[cb + 16] * v1;
      }
    }
  }
}

// ------------------------------------------------------------- workspace layout
constexpr size_t algn(size_t x) { return (x + 255) & ~(size_t)255; }
constexpr size_t OFF_PERM = 0;
constexpr size_t OFF_META = algn(OFF_PERM + (size_t)CAP * 4);
constexpr size_t OFF_XM   = algn(OFF_META + 4096);
constexpr size_t OFF_XS   = algn(OFF_XM   + (size_t)NTOK * DIM * 4);
constexpr size_t OFF_XG   = algn(OFF_XS   + (size_t)NTOK * DIM * 2);
constexpr size_t OFF_UG   = algn(OFF_XG   + (size_t)CAP * DIM * 2);
constexpr size_t OFF_SH   = algn(OFF_UG   + (size_t)CAP * STATE * 2);
constexpr size_t OFF_SEL  = algn(OFF_SH   + (size_t)CAP * SELH * 2);
constexpr size_t OFF_YG   = algn(OFF_SEL  + (size_t)CAP * 4 * STATE * 2);
constexpr size_t OFF_HACT = algn(OFF_YG   + (size_t)CAP * STATE * 2);
constexpr size_t OFF_WIN  = algn(OFF_HACT + (size_t)NTOK * 4 * DIM * 2);
constexpr size_t OFF_WSI  = algn(OFF_WIN  + (size_t)NE * DIM * STATE * 2);
constexpr size_t OFF_WSO  = algn(OFF_WSI  + (size_t)NE * DIM * SELH * 2);
constexpr size_t OFF_WOUT = algn(OFF_WSO  + (size_t)NE * SELH * 4 * STATE * 2);
constexpr size_t OFF_WFC  = algn(OFF_WOUT + (size_t)NE * STATE * DIM * 2);
constexpr size_t OFF_WPJ  = algn(OFF_WFC  + (size_t)DIM * 4 * DIM * 2);

extern "C" void kernel_launch(void* const* d_in, const int* in_sizes, int n_in,
                              void* d_out, int out_size, void* d_ws, size_t ws_size,
                              hipStream_t stream) {
  (void)in_sizes; (void)n_in; (void)out_size; (void)ws_size;
  const float* x     = (const float*)d_in[0];
  const float* x0    = (const float*)d_in[1];
  const float* W_in  = (const float*)d_in[2];
  const float* W_si  = (const float*)d_in[3];
  const float* W_so  = (const float*)d_in[4];
  const float* W_out = (const float*)d_in[5];
  const float* d_par = (const float*)d_in[6];
  const float* ssm_s = (const float*)d_in[7];
  const float* mlp_s = (const float*)d_in[8];
  const float* mix   = (const float*)d_in[9];
  const float* W_fc  = (const float*)d_in[10];
  const float* W_pj  = (const float*)d_in[11];
  const int*   tok   = (const int*)d_in[12];
  float* out = (float*)d_out;
  char*  ws  = (char*)d_ws;

  int*   perm = (int*)(ws + OFF_PERM);
  int*   meta = (int*)(ws + OFF_META);
  float* xm   = (float*)(ws + OFF_XM);
  u16*   xs   = (u16*)(ws + OFF_XS);
  u16*   Xg   = (u16*)(ws + OFF_XG);
  u16*   Ug   = (u16*)(ws + OFF_UG);
  u16*   Sh   = (u16*)(ws + OFF_SH);
  u16*   Sel  = (u16*)(ws + OFF_SEL);
  u16*   Yg   = (u16*)(ws + OFF_YG);
  u16*   Ha   = (u16*)(ws + OFF_HACT);
  u16*   bWin = (u16*)(ws + OFF_WIN);
  u16*   bWsi = (u16*)(ws + OFF_WSI);
  u16*   bWso = (u16*)(ws + OFF_WSO);
  u16*   bWou = (u16*)(ws + OFF_WOUT);
  u16*   bWfc = (u16*)(ws + OFF_WFC);
  u16*   bWpj = (u16*)(ws + OFF_WPJ);

  cvt_bf16<<<1024, 256, 0, stream>>>(W_in,  bWin, NE * DIM * STATE);
  cvt_bf16<<<1024, 256, 0, stream>>>(W_si,  bWsi, NE * DIM * SELH);
  cvt_bf16<<<1024, 256, 0, stream>>>(W_so,  bWso, NE * SELH * 4 * STATE);
  cvt_bf16<<<1024, 256, 0, stream>>>(W_out, bWou, NE * STATE * DIM);
  cvt_bf16<<<1024, 256, 0, stream>>>(W_fc,  bWfc, DIM * 4 * DIM);
  cvt_bf16<<<1024, 256, 0, stream>>>(W_pj,  bWpj, 4 * DIM * DIM);

  route_build<<<1, 256, 0, stream>>>(tok, perm, meta);
  mix_rmsnorm<<<NTOK, 256, 0, stream>>>(x, x0, mix, xm, xs);
  gather_rows<<<CAP, 256, 0, stream>>>(perm, xs, Xg);

  // routed SSM projections (grouped WMMA GEMMs)
  gemm_wmma<0><<<dim3(STATE / 64, MT), 256, 0, stream>>>(
      Xg, DIM, CAP, bWin, STATE, DIM, (long long)DIM * STATE, meta + 17,
      nullptr, Ug, nullptr, nullptr, nullptr);
  gemm_wmma<1><<<dim3(SELH / 64, MT), 256, 0, stream>>>(
      Xg, DIM, CAP, bWsi, SELH, DIM, (long long)DIM * SELH, meta + 17,
      nullptr, Sh, nullptr, nullptr, nullptr);
  gemm_wmma<2><<<dim3(4 * STATE / 64, MT), 256, 0, stream>>>(
      Sh, SELH, CAP, bWso, 4 * STATE, SELH, (long long)SELH * 4 * STATE, meta + 17,
      nullptr, Sel, nullptr, nullptr, nullptr);

  ssm_scan<<<NGRP * 4, 256, 0, stream>>>(Ug, Sel, d_par, meta, Yg);

  // routed out-proj, fused x2 = xm + ssm_scale * ssm_out (scatter by perm)
  gemm_wmma<3><<<dim3(DIM / 64, MT), 256, 0, stream>>>(
      Yg, STATE, CAP, bWou, DIM, STATE, (long long)STATE * DIM, meta + 17,
      perm, nullptr, nullptr, ssm_s, xm);

  rmsnorm2<<<NTOK, 256, 0, stream>>>(xm, xs);

  // MLP: relu^2(xn @ W_fc) @ W_proj, fused final residual
  gemm_wmma<4><<<dim3(4 * DIM / 64, NTOK / 64), 256, 0, stream>>>(
      xs, DIM, NTOK, bWfc, 4 * DIM, DIM, 0, nullptr,
      nullptr, Ha, nullptr, nullptr, nullptr);
  gemm_wmma<5><<<dim3(DIM / 64, NTOK / 64), 256, 0, stream>>>(
      Ha, 4 * DIM, NTOK, bWpj, DIM, 4 * DIM, 0, nullptr,
      nullptr, nullptr, out, mlp_s, xm);
}
